// MessagePassingLayer_28217935135436
// MI455X (gfx1250) — compile-verified, matrix operand order checked by reading, not run on hardware
//
#include <hip/hip_runtime.h>

typedef _Float16 half_t;
typedef __attribute__((ext_vector_type(16))) _Float16 v16h;
typedef __attribute__((ext_vector_type(2)))  _Float16 h2;
typedef __attribute__((ext_vector_type(8)))  float    v8f;

#define B_   8
#define N_   256
#define F_   64
#define JS   8            // j slices
#define JPS  (N_ / JS)    // 32 j per slice
#define LN_EPS 1e-5f

// ---------------------------------------------------------------------------
// Stage 0a: Wec = e_w2 @ W1c (64x64), cv = m_b1 + e_b2 @ W1c
// ---------------------------------------------------------------------------
__global__ __launch_bounds__(256) void k0_wec(
    const float* __restrict__ e_w2, const float* __restrict__ e_b2,
    const float* __restrict__ m_w1, const float* __restrict__ m_b1,
    float* __restrict__ Wec, float* __restrict__ cv)
{
    int tid = threadIdx.x;
    for (int idx = tid; idx < 64 * 64; idx += 256) {
        int k = idx >> 6, c = idx & 63;
        float acc = 0.0f;
        for (int t = 0; t < 64; ++t)
            acc = fmaf(e_w2[k * 64 + t], m_w1[(128 + t) * 64 + c], acc);
        Wec[idx] = acc;
    }
    if (tid < 64) {
        float acc = m_b1[tid];
        for (int t = 0; t < 64; ++t)
            acc = fmaf(e_b2[t], m_w1[(128 + t) * 64 + tid], acc);
        cv[tid] = acc;
    }
}

// ---------------------------------------------------------------------------
// Stage 0b: pack Wec (f32, row-major [k][c]) into f16 WMMA B-operand layout.
// Layout: wecB[nt][kh][r][lane] (u32 = half2). lane -> n = lane&15;
// reg r holds K pair kbase = (r<4 ? 2r : 16+2(r-4)) + (lane>=16 ? 8 : 0),
// within K-half kh (kh*32 offset).
// ---------------------------------------------------------------------------
__global__ __launch_bounds__(256) void k0_pack(
    const float* __restrict__ Wec, unsigned* __restrict__ wecB)
{
    int idx  = blockIdx.x * 256 + threadIdx.x;   // 0..2047
    int lane = idx & 31;
    int r    = (idx >> 5) & 7;
    int kh   = (idx >> 8) & 1;
    int nt   = idx >> 9;
    int n    = lane & 15;
    int koff = (r < 4 ? 2 * r : 16 + 2 * (r - 4)) + ((lane >= 16) ? 8 : 0);
    int k    = kh * 32 + koff;
    h2 p;
    p.x = (half_t)Wec[(k    ) * 64 + nt * 16 + n];
    p.y = (half_t)Wec[(k + 1) * 64 + nt * 16 + n];
    wecB[idx] = __builtin_bit_cast(unsigned, p);
}

// ---------------------------------------------------------------------------
// Stage 0c: Apre = nf @ W1a + cv  ;  Bpre = nf @ W1b     (B,N,64 each, f32)
// ---------------------------------------------------------------------------
__global__ __launch_bounds__(256) void k0_ab(
    const float* __restrict__ nf, const float* __restrict__ m_w1,
    const float* __restrict__ cv,
    float* __restrict__ Apre, float* __restrict__ Bpre)
{
    int idx = blockIdx.x * 256 + threadIdx.x;    // b*N*F elements
    int c   = idx & 63;
    int bn  = idx >> 6;
    const float* x = nf + bn * 64;
    float a = cv[c], bb = 0.0f;
    for (int k = 0; k < 64; ++k) {
        float xk = x[k];
        a  = fmaf(xk, m_w1[k * 64 + c],        a);
        bb = fmaf(xk, m_w1[(64 + k) * 64 + c], bb);
    }
    Apre[idx] = a;
    Bpre[idx] = bb;
}

// ---------------------------------------------------------------------------
// Stage 1: the N^2 pair loop. Block = (b, jslice), 16 waves = 16 i-tiles.
// Per wave: S[16 i][64 ch] += w_ij * relu( WMMA(relu(d*w+b), Wec) + Apre + Bpre )
// ---------------------------------------------------------------------------
__global__ __launch_bounds__(512) void stage1(
    const float* __restrict__ pos, const float* __restrict__ cell_len,
    const float* __restrict__ mask,
    const float* __restrict__ e_w1, const float* __restrict__ e_b1,
    const float* __restrict__ Apre, const float* __restrict__ Bpre,
    const unsigned* __restrict__ wecB,
    float* __restrict__ Spart, float* __restrict__ cntpart)
{
    __shared__ float sh_pos[N_];
    __shared__ float sh_mj[JPS];
    __shared__ float sh_Bpre[JPS * F_];

    const int b   = blockIdx.x / JS;
    const int js  = blockIdx.x % JS;
    const int tid = threadIdx.x;

    for (int t = tid; t < N_; t += 512)       sh_pos[t]  = pos[b * N_ + t];
    for (int t = tid; t < JPS; t += 512)      sh_mj[t]   = mask[b * N_ + js * JPS + t];
    for (int t = tid; t < JPS * F_; t += 512) sh_Bpre[t] = Bpre[(b * N_ + js * JPS) * F_ + t];
    __syncthreads();

    const int wave = tid >> 5;        // i-tile
    const int lane = tid & 31;
    const int i0   = wave * 16;
    const int m    = lane & 15;
    const int hi   = lane >> 4;       // 0/1

    const float pi       = sh_pos[i0 + m];
    const float cell     = cell_len[b];
    const float inv_cell = 1.0f / cell;

    // Apre tile in D layout (reg t -> row i0+t+8*hi, lane -> channel nt*16+m)
    float Ap[4][8];
#pragma unroll
    for (int nt = 0; nt < 4; ++nt)
#pragma unroll
        for (int t = 0; t < 8; ++t)
            Ap[nt][t] = Apre[(b * N_ + i0 + t + hi * 8) * F_ + nt * 16 + m];

    // e_w1/e_b1 packed per-lane in A-operand order
    h2 wp[2][8], bp[2][8];
#pragma unroll
    for (int kh = 0; kh < 2; ++kh)
#pragma unroll
        for (int r = 0; r < 8; ++r) {
            int koff = (r < 4 ? 2 * r : 16 + 2 * (r - 4)) + hi * 8;
            int k    = kh * 32 + koff;
            h2 w, bb;
            w.x = (half_t)e_w1[k];   w.y = (half_t)e_w1[k + 1];
            bb.x = (half_t)e_b1[k];  bb.y = (half_t)e_b1[k + 1];
            wp[kh][r] = w;  bp[kh][r] = bb;
        }

    // Wec as resident B operands: Bop[nt][kh] (v16h each)
    v16h Bop[4][2];
#pragma unroll
    for (int nt = 0; nt < 4; ++nt)
#pragma unroll
        for (int kh = 0; kh < 2; ++kh)
#pragma unroll
            for (int r = 0; r < 8; ++r) {
                unsigned u = wecB[(nt * 2 + kh) * 256 + r * 32 + lane];
                h2 p = __builtin_bit_cast(h2, u);
                Bop[nt][kh][2 * r]     = p.x;
                Bop[nt][kh][2 * r + 1] = p.y;
            }

    v8f S[4];
#pragma unroll
    for (int nt = 0; nt < 4; ++nt) S[nt] = (v8f){0, 0, 0, 0, 0, 0, 0, 0};
    float cnt[8];
#pragma unroll
    for (int t = 0; t < 8; ++t) cnt[t] = 0.0f;

    const h2 hz = {(half_t)0.0f, (half_t)0.0f};

    for (int jj = 0; jj < JPS; ++jj) {
        const int   j  = js * JPS + jj;
        const float pj = sh_pos[j];
        const float mj = sh_mj[jj];

        float d    = fabsf(pi - pj);
        float dist = fminf(d, cell - d) * inv_cell;
        half_t dh  = (half_t)dist;
        h2 d2      = {dh, dh};

        // Build A operand: relu(dist * e_w1 + e_b1), packed f16
        v16h A0, A1;
#pragma unroll
        for (int r = 0; r < 8; ++r) {
            h2 t0 = d2 * wp[0][r] + bp[0][r];
            t0 = __builtin_elementwise_max(t0, hz);
            A0[2 * r] = t0.x;  A0[2 * r + 1] = t0.y;
            h2 t1 = d2 * wp[1][r] + bp[1][r];
            t1 = __builtin_elementwise_max(t1, hz);
            A1[2 * r] = t1.x;  A1[2 * r + 1] = t1.y;
        }

        float wfac[8];
#pragma unroll
        for (int t = 0; t < 8; ++t) {
            wfac[t] = ((i0 + t + hi * 8) == j) ? 0.0f : mj;
            cnt[t] += wfac[t];
        }

#pragma unroll
        for (int nt = 0; nt < 4; ++nt) {
            v8f acc = (v8f){0, 0, 0, 0, 0, 0, 0, 0};
            acc = __builtin_amdgcn_wmma_f32_16x16x32_f16(
                      false, A0, false, Bop[nt][0], (short)0, acc, false, false);
            acc = __builtin_amdgcn_wmma_f32_16x16x32_f16(
                      false, A1, false, Bop[nt][1], (short)0, acc, false, false);
            const float Bj = sh_Bpre[jj * F_ + nt * 16 + m];
#pragma unroll
            for (int t = 0; t < 8; ++t) {
                float h = acc[t] + Ap[nt][t] + Bj;
                h = fmaxf(h, 0.0f);
                S[nt][t] = fmaf(h, wfac[t], S[nt][t]);
            }
        }
    }

    // store partials
#pragma unroll
    for (int nt = 0; nt < 4; ++nt)
#pragma unroll
        for (int t = 0; t < 8; ++t) {
            int i = i0 + t + hi * 8;
            Spart[(((b * JS) + js) * N_ + i) * F_ + nt * 16 + m] = S[nt][t];
        }
    if (m == 0) {
#pragma unroll
        for (int t = 0; t < 8; ++t)
            cntpart[((b * JS) + js) * N_ + i0 + t + hi * 8] = cnt[t];
    }
}

// ---------------------------------------------------------------------------
// Stage 2: per node — reduce partials, agg = mi*(S@m_w2 + cnt*m_b2),
// update MLP, residual, LayerNorm, mask. One wave (32 lanes) per node,
// each lane owns channels (lane, lane+32).
// ---------------------------------------------------------------------------
__global__ __launch_bounds__(256) void stage2(
    const float* __restrict__ Spart, const float* __restrict__ cntpart,
    const float* __restrict__ nf, const float* __restrict__ mask,
    const float* __restrict__ m_w2, const float* __restrict__ m_b2,
    const float* __restrict__ u_w1, const float* __restrict__ u_b1,
    const float* __restrict__ u_w2, const float* __restrict__ u_b2,
    const float* __restrict__ ln_g, const float* __restrict__ ln_b,
    float* __restrict__ out)
{
    __shared__ float shm[8][64];
    __shared__ float shx[8][64];

    const int wave = threadIdx.x >> 5;
    const int lane = threadIdx.x & 31;
    const int node = blockIdx.x * 8 + wave;
    const int b    = node / N_;
    const int i    = node % N_;

    float S0 = 0.0f, S1 = 0.0f, cnt = 0.0f;
    for (int js = 0; js < JS; ++js) {
        const float* p = Spart + ((b * JS + js) * N_ + i) * F_;
        S0 += p[lane];
        S1 += p[lane + 32];
        cnt += cntpart[(b * JS + js) * N_ + i];
    }

    const float x0 = nf[(b * N_ + i) * F_ + lane];
    const float x1 = nf[(b * N_ + i) * F_ + lane + 32];
    shm[wave][lane] = S0;  shm[wave][lane + 32] = S1;
    shx[wave][lane] = x0;  shx[wave][lane + 32] = x1;
    __syncthreads();

    const float mi = mask[b * N_ + i];

    // agg = mi * (S @ m_w2 + cnt * m_b2)
    float a0 = cnt * m_b2[lane], a1 = cnt * m_b2[lane + 32];
    for (int k = 0; k < 64; ++k) {
        float s = shm[wave][k];
        a0 = fmaf(s, m_w2[k * 64 + lane],      a0);
        a1 = fmaf(s, m_w2[k * 64 + lane + 32], a1);
    }
    a0 *= mi;  a1 *= mi;
    __syncthreads();
    shm[wave][lane] = a0;  shm[wave][lane + 32] = a1;
    __syncthreads();

    // h = relu([x, agg] @ u_w1 + u_b1)
    float h0 = u_b1[lane], h1 = u_b1[lane + 32];
    for (int k = 0; k < 64; ++k) {
        float xk = shx[wave][k], ak = shm[wave][k];
        h0 = fmaf(xk, u_w1[k * 64 + lane],             h0);
        h1 = fmaf(xk, u_w1[k * 64 + lane + 32],        h1);
        h0 = fmaf(ak, u_w1[(64 + k) * 64 + lane],      h0);
        h1 = fmaf(ak, u_w1[(64 + k) * 64 + lane + 32], h1);
    }
    h0 = fmaxf(h0, 0.0f);  h1 = fmaxf(h1, 0.0f);
    __syncthreads();
    shm[wave][lane] = h0;  shm[wave][lane + 32] = h1;
    __syncthreads();

    // upd = h @ u_w2 + u_b2 ; residual
    float u0 = u_b2[lane], u1 = u_b2[lane + 32];
    for (int k = 0; k < 64; ++k) {
        float hk = shm[wave][k];
        u0 = fmaf(hk, u_w2[k * 64 + lane],      u0);
        u1 = fmaf(hk, u_w2[k * 64 + lane + 32], u1);
    }
    float y0 = x0 + u0, y1 = x1 + u1;

    // LayerNorm over 64 channels (wave32 xor-shuffle reduction)
    float s  = y0 + y1;
    float sq = y0 * y0 + y1 * y1;
    for (int off = 16; off > 0; off >>= 1) {
        s  += __shfl_xor(s,  off, 32);
        sq += __shfl_xor(sq, off, 32);
    }
    float mu  = s * (1.0f / 64.0f);
    float var = sq * (1.0f / 64.0f) - mu * mu;
    float r   = rsqrtf(var + LN_EPS);

    out[(b * N_ + i) * F_ + lane]      = ((y0 - mu) * r * ln_g[lane]      + ln_b[lane])      * mi;
    out[(b * N_ + i) * F_ + lane + 32] = ((y1 - mu) * r * ln_g[lane + 32] + ln_b[lane + 32]) * mi;
}

// ---------------------------------------------------------------------------
extern "C" void kernel_launch(void* const* d_in, const int* in_sizes, int n_in,
                              void* d_out, int out_size, void* d_ws, size_t ws_size,
                              hipStream_t stream)
{
    const float* nf   = (const float*)d_in[0];
    const float* pos  = (const float*)d_in[1];
    const float* cl   = (const float*)d_in[2];
    const float* mask = (const float*)d_in[3];
    const float* e_w1 = (const float*)d_in[4];
    const float* e_b1 = (const float*)d_in[5];
    const float* e_w2 = (const float*)d_in[6];
    const float* e_b2 = (const float*)d_in[7];
    const float* m_w1 = (const float*)d_in[8];
    const float* m_b1 = (const float*)d_in[9];
    const float* m_w2 = (const float*)d_in[10];
    const float* m_b2 = (const float*)d_in[11];
    const float* u_w1 = (const float*)d_in[12];
    const float* u_b1 = (const float*)d_in[13];
    const float* u_w2 = (const float*)d_in[14];
    const float* u_b2 = (const float*)d_in[15];
    const float* ln_g = (const float*)d_in[16];
    const float* ln_b = (const float*)d_in[17];

    char* ws = (char*)d_ws;
    float*    Spart   = (float*)(ws + 0);            // 8*8*256*64*4 = 4 MB
    float*    cntpart = (float*)(ws + 4194304);      // 64 KB
    float*    Apre    = (float*)(ws + 4259840);      // 512 KB
    float*    Bpre    = (float*)(ws + 4784128);      // 512 KB
    float*    Wec     = (float*)(ws + 5308416);      // 16 KB
    float*    cv      = (float*)(ws + 5324800);      // 256 B
    unsigned* wecB    = (unsigned*)(ws + 5325056);   // 8 KB

    k0_wec <<<1,   256, 0, stream>>>(e_w2, e_b2, m_w1, m_b1, Wec, cv);
    k0_pack<<<8,   256, 0, stream>>>(Wec, wecB);
    k0_ab  <<<(B_ * N_ * F_) / 256, 256, 0, stream>>>(nf, m_w1, cv, Apre, Bpre);
    stage1 <<<B_ * JS, 512, 0, stream>>>(pos, cl, mask, e_w1, e_b1,
                                         Apre, Bpre, wecB, Spart, cntpart);
    stage2 <<<(B_ * N_) / 8, 256, 0, stream>>>(Spart, cntpart, nf, mask,
                                               m_w2, m_b2, u_w1, u_b1,
                                               u_w2, u_b2, ln_g, ln_b,
                                               (float*)d_out);
}
